// Criterion_8761733284570
// MI455X (gfx1250) — matrix operand order repaired and checked by reading, not compiled, not run on hardware
//
#include <hip/hip_runtime.h>
#include <math.h>

// ---------------- CDNA5 WMMA types ----------------
typedef __attribute__((ext_vector_type(16))) __bf16 v16bf;
typedef __attribute__((ext_vector_type(8)))  float  v8f;

#define EPS_W   0.05f
#define MAX_IT  100
#define THRESH  0.1f
#define MARGIN  0.2f
#define CC      128          // channels
#define HW      256          // spatial positions
#define NPAIR   512          // 2 * N_TRIPLETS

// LDS layout (bytes). Rows padded to avoid bank conflicts.
#define ROWB   260           // 130 bf16 per row (128 used) -> 65 dwords (conflict-free)
#define XN_OFF 0
#define YN_OFF (XN_OFF + HW*ROWB)            // 66560
#define K_OFF  (YN_OFF + HW*ROWB)            // 133120
#define KROWB  516                           // 258 bf16 per row (256 used) -> 129 dwords
#define VEC_OFF (K_OFF + HW*KROWB)           // 265216
// float arrays at VEC_OFF: xm[128] ym[128] u[256] v[256] r[256] c[256] scalars[8]
#define SMEM_BYTES (VEC_OFF + (128+128+256+256+256+256+8)*4)

__device__ __forceinline__ unsigned short f2bf(float f) {
    unsigned u = __builtin_bit_cast(unsigned, f);
    u += 0x7FFFu + ((u >> 16) & 1u);          // round-to-nearest-even
    return (unsigned short)(u >> 16);
}
__device__ __forceinline__ float bf2f(unsigned short h) {
    unsigned u = ((unsigned)h) << 16;
    return __builtin_bit_cast(float, u);
}

__global__ void __launch_bounds__(256)
wdist_pair_kernel(const float* __restrict__ batch,
                  const int*   __restrict__ triplets,
                  float*       __restrict__ dpair)
{
    extern __shared__ unsigned char smem[];
    float* fp   = (float*)(smem + VEC_OFF);
    float* xm_s = fp;            // 128
    float* ym_s = fp + 128;      // 128
    float* u_s  = fp + 256;      // 256
    float* v_s  = fp + 512;      // 256
    float* r_s  = fp + 768;      // 256
    float* c_s  = fp + 1024;     // 256
    float* sc_s = fp + 1280;     // [0]=usum [1]=vsum [2]=err [3]=d1 [4]=dist2

    const int t    = threadIdx.x;       // 0..255
    const int lane = t & 31;
    const int wid  = t >> 5;            // 8 waves
    const int half = lane >> 4;
    const int l16  = lane & 15;

    const int pid  = blockIdx.x;
    const int tri  = pid & 255;
    const int isng = pid >> 8;
    const int ia   = triplets[3*tri + 0];
    const int ib   = triplets[3*tri + 1 + isng];
    const float* x = batch + (size_t)ia * CC * HW;
    const float* y = batch + (size_t)ib * CC * HW;

    // ---- Phase 1: per-column L2 normalize (across C), store transposed bf16 in LDS.
    {
        float ssx = 0.f, ssy = 0.f;
        for (int c = 0; c < CC; ++c) {
            float vx = x[c*HW + t]; ssx += vx*vx;
            float vy = y[c*HW + t]; ssy += vy*vy;
        }
        float sx = 1.f / fmaxf(sqrtf(ssx), 1e-12f);
        float sy = 1.f / fmaxf(sqrtf(ssy), 1e-12f);
        unsigned char* xrow = smem + XN_OFF + t*ROWB;
        unsigned char* yrow = smem + YN_OFF + t*ROWB;
        for (int c = 0; c < CC; ++c) {
            *(unsigned short*)(xrow + 2*c) = f2bf(x[c*HW + t] * sx);
            *(unsigned short*)(yrow + 2*c) = f2bf(y[c*HW + t] * sy);
        }
        // per-channel means (threads 0..127 -> x, 128..255 -> y)
        const float* src = (t < 128) ? x : y;
        int c = t & 127;
        float acc = 0.f;
        for (int s = 0; s < HW; ++s) acc += src[c*HW + s];
        if (t < 128) xm_s[c] = acc * (1.0f/HW); else ym_s[c] = acc * (1.0f/HW);
    }
    __syncthreads();

    // ---- Phase 2: normalize mean vectors (serial, tiny), init scalars.
    if (t == 0) {
        float n = 0.f; for (int c = 0; c < CC; ++c) n += xm_s[c]*xm_s[c];
        float s = 1.f / fmaxf(sqrtf(n), 1e-12f);
        for (int c = 0; c < CC; ++c) xm_s[c] *= s;
    } else if (t == 1) {
        float n = 0.f; for (int c = 0; c < CC; ++c) n += ym_s[c]*ym_s[c];
        float s = 1.f / fmaxf(sqrtf(n), 1e-12f);
        for (int c = 0; c < CC; ++c) ym_s[c] *= s;
    } else if (t == 2) {
        sc_s[0] = 0.f; sc_s[1] = 0.f; sc_s[3] = 0.f;
    }
    __syncthreads();

    // ---- Phase 3: attention weights u,v; dist2.
    {
        const unsigned short* yrow = (const unsigned short*)(smem + YN_OFF + t*ROWB);
        const unsigned short* xrow = (const unsigned short*)(smem + XN_OFF + t*ROWB);
        float au = 0.f, av = 0.f;
        for (int c = 0; c < CC; ++c) {
            au += xm_s[c] * bf2f(yrow[c]);
            av += ym_s[c] * bf2f(xrow[c]);
        }
        au = fmaxf(au, 0.f); av = fmaxf(av, 0.f);
        u_s[t] = au; v_s[t] = av;
        atomicAdd(&sc_s[0], au);
        atomicAdd(&sc_s[1], av);
        if (t == 0) {
            float d = 0.f; for (int c = 0; c < CC; ++c) d += xm_s[c]*ym_s[c];
            sc_s[4] = sqrtf(fmaxf(2.f - 2.f*d, 0.f) + 1e-6f);
        }
    }
    __syncthreads();
    u_s[t] = u_s[t] / (sc_s[0] + 1e-5f);
    v_s[t] = v_s[t] / (sc_s[1] + 1e-5f);
    r_s[t] = 1.f; c_s[t] = 1.f;
    __syncthreads();

    // ---- Phase 4: sim = xn^T * yn via WMMA bf16 -> f32, then K = exp((sim-1)/eps) into LDS.
    // 16x16 tile grid (16x16 = 256 tiles), 8 waves * 32 tiles each.
    for (int i = 0; i < 32; ++i) {
        int tile = wid * 32 + i;
        int tm = tile >> 4, tn = tile & 15;
        v8f cfrag = {};
        const unsigned char* arow = smem + XN_OFF + (tm*16 + l16)*ROWB;
        const unsigned char* brow = smem + YN_OFF + (tn*16 + l16)*ROWB;
        #pragma unroll
        for (int kk = 0; kk < 4; ++kk) {
            union { v16bf v; unsigned u[8]; } A, B;
            #pragma unroll
            for (int vv = 0; vv < 8; ++vv) {
                int ka = ((vv < 4) ? 2*vv : 16 + 2*(vv-4)) + 8*half + 32*kk;  // A layout
                A.u[vv] = *(const unsigned*)(arow + 2*ka);
                int kb = ((vv < 4) ? 2*vv : 16 + 2*(vv-4)) + 8*half + 32*kk;  // B layout
                B.u[vv] = *(const unsigned*)(brow + 2*kb);
            }
            cfrag = __builtin_amdgcn_wmma_f32_16x16x32_bf16(
                        false, A.v, false, B.v, (short)0, cfrag, false, false);
        }
        int n = tn*16 + l16;
        #pragma unroll
        for (int vv = 0; vv < 8; ++vv) {
            int m = tm*16 + vv + 8*half;                                       // C/D layout
            float kv = expf((cfrag[vv] - 1.0f) * (1.0f/EPS_W));
            *(unsigned short*)(smem + K_OFF + m*KROWB + 2*n) = f2bf(kv);
        }
    }
    __syncthreads();

    // ---- Phase 5: Sinkhorn (workgroup-uniform loop; thread t owns row/col t).
    float err = 1e9f;
    int it = 0;
    while (it < MAX_IT && err >= THRESH) {
        // r_new = u / (K @ c)
        const unsigned* krow = (const unsigned*)(smem + K_OFF + t*KROWB);
        float dot = 0.f;
        for (int j = 0; j < 128; ++j) {
            unsigned w = krow[j];
            dot += bf2f((unsigned short)(w & 0xFFFFu)) * c_s[2*j]
                 + bf2f((unsigned short)(w >> 16))     * c_s[2*j + 1];
        }
        float rn   = u_s[t] / dot;
        float diff = fabsf(rn - r_s[t]);
        __syncthreads();
        if (t == 0) sc_s[2] = 0.f;
        __syncthreads();
        atomicAdd(&sc_s[2], diff);
        r_s[t] = rn;
        __syncthreads();
        // c_new = v / (K^T @ r)
        float dot2 = 0.f;
        for (int m = 0; m < HW; ++m)
            dot2 += bf2f(*(const unsigned short*)(smem + K_OFF + m*KROWB + 2*t)) * r_s[m];
        float cn = v_s[t] / dot2;
        err = sc_s[2] * (1.0f/HW);
        __syncthreads();
        c_s[t] = cn;
        __syncthreads();
        ++it;
    }

    // ---- Phase 6: d1 = sum T*dist1, T = r c K, dist1 = sqrt(max(-2*eps*lnK,0)+1e-6).
    {
        const unsigned short* kr = (const unsigned short*)(smem + K_OFF + t*KROWB);
        float rt = r_s[t], acc = 0.f;
        for (int n = 0; n < HW; ++n) {
            float kb = bf2f(kr[n]);
            float d1 = sqrtf(fmaxf(-2.0f*EPS_W*logf(kb), 0.f) + 1e-6f);
            acc += rt * c_s[n] * kb * d1;
        }
        atomicAdd(&sc_s[3], acc);
    }
    __syncthreads();
    if (t == 0) dpair[pid] = (sc_s[3] + sc_s[4]) * 0.5f;
}

__global__ void __launch_bounds__(256)
triplet_loss_kernel(const float* __restrict__ dpair,
                    const float* __restrict__ beta,
                    const int*   __restrict__ labels,
                    const int*   __restrict__ triplets,
                    float*       __restrict__ out)
{
    __shared__ float tot_s, cnt_s;
    int t = threadIdx.x;
    if (t == 0) { tot_s = 0.f; cnt_s = 0.f; }
    __syncthreads();
    float d_ap = dpair[t];
    float d_an = dpair[256 + t];
    int   a    = triplets[3*t];
    float b    = beta[labels[a]];
    float pos  = fmaxf(d_ap - b + MARGIN, 0.f);
    float neg  = fmaxf(b - d_an + MARGIN, 0.f);
    float cnt  = (pos > 0.f ? 1.f : 0.f) + (neg > 0.f ? 1.f : 0.f);
    atomicAdd(&tot_s, pos + neg);
    atomicAdd(&cnt_s, cnt);
    __syncthreads();
    if (t == 0) out[0] = (cnt_s == 0.f) ? tot_s : (tot_s / cnt_s);
}

extern "C" void kernel_launch(void* const* d_in, const int* in_sizes, int n_in,
                              void* d_out, int out_size, void* d_ws, size_t ws_size,
                              hipStream_t stream) {
    const float* batch    = (const float*)d_in[0];
    const float* beta     = (const float*)d_in[1];
    const int*   labels   = (const int*)d_in[2];
    const int*   triplets = (const int*)d_in[3];
    float*       dpair    = (float*)d_ws;          // 512 floats
    float*       out      = (float*)d_out;

    // Allow >64KB dynamic LDS (270 KB of the WGP's 320 KB).
    (void)hipFuncSetAttribute((const void*)wdist_pair_kernel,
                              hipFuncAttributeMaxDynamicSharedMemorySize, SMEM_BYTES);

    wdist_pair_kernel<<<NPAIR, 256, SMEM_BYTES, stream>>>(batch, triplets, dpair);
    triplet_loss_kernel<<<1, 256, 0, stream>>>(dpair, beta, labels, triplets, out);
}
